// FlashBlocksparseAttention_32255204393438
// MI455X (gfx1250) — matrix operand order repaired and checked by reading, not compiled, not run on hardware
//
#include <hip/hip_runtime.h>

typedef __attribute__((ext_vector_type(16))) _Float16 v16h;
typedef __attribute__((ext_vector_type(8)))  _Float16 v8h;
typedef __attribute__((ext_vector_type(8)))  float    v8f;

#define B_   2
#define S_   2048
#define H_   16
#define D_   64
#define BLKQ 16
#define BLKK 256
#define NK   (S_ / BLKK)      // 8 key blocks
#define NQ   (S_ / BLKQ)      // 128 query blocks
#define SROW (3 * H_ * D_)    // 3072 floats between sequence positions
#define WAVES 16              // waves per workgroup == q-blocks per group
#define TPB   (WAVES * 32)    // 512 threads

union HV16 { v16h v; v8h h[2]; };

// bare v_max_num_f32 (IEEE maximumNumber) -- avoids fmaxf's extra
// quiet-NaN canonicalization ops.
__device__ __forceinline__ float vmaxf(float a, float b) {
  float d;
  asm("v_max_num_f32 %0, %1, %2" : "=v"(d) : "v"(a), "v"(b));
  return d;
}

// fused reduction step: d = max(row_xmask<MASK>(x), x) in a single VOP2+DPP.
#define ROWMAX_STEP(x, MASK)                                                   \
  { float _d;                                                                  \
    asm("v_max_num_f32_dpp %0, %1, %2 row_xmask:" #MASK                        \
        " row_mask:0xf bank_mask:0xf bound_ctrl:1"                             \
        : "=v"(_d) : "v"(x), "v"(x));                                          \
    (x) = _d; }

// max across each 16-lane row (lanes 0-15 / 16-31 independently)
__device__ __forceinline__ float rowmax16(float x) {
  ROWMAX_STEP(x, 1)
  ROWMAX_STEP(x, 2)
  ROWMAX_STEP(x, 4)
  ROWMAX_STEP(x, 8)
  return x;
}

__global__ __launch_bounds__(TPB)
void fa_blocksparse_wmma_coop(const float* __restrict__ qkv,
                              const int*   __restrict__ layout,
                              float*       __restrict__ out)
{
  // Double-buffered, workgroup-shared staged chunks (f16), 16B-aligned so
  // fragment reads lower to ds_load_b128.
  __shared__ __align__(16) _Float16 klds[2][32 * 64];      // [buf][key][d]
  __shared__ __align__(16) _Float16 vtld[2][64 * 32];      // [buf][d][key] transposed
  __shared__ __align__(16) _Float16 plds[WAVES][16 * 32];  // per-wave P relayout

  const int tid  = threadIdx.x;      // 0..511
  const int lane = tid & 31;
  const int w    = tid >> 5;         // wave id == q-block within group
  const int lr   = lane & 15;
  const int hi   = lane >> 4;
  const int g    = blockIdx.x;       // q-block group of 16
  const int h    = blockIdx.y;
  const int b    = blockIdx.z;
  const int qb   = g * 16 + w;
  const int q0   = qb * BLKQ;

  const float* base = qkv + (size_t)b * S_ * SROW;
  const float* qp = base + 0 * H_ * D_ + h * D_;
  const float* kp = base + 1 * H_ * D_ + h * D_;
  const float* vp = base + 2 * H_ * D_ + h * D_;

  // ---- Q tile (16x64) -> two 16x32 f16 A-fragments ----
  // Pre-scale by log2(e)/sqrt(D): scores land in the log2 domain, so
  // softmax uses raw v_exp_f32 (base-2) with no extra multiplies.
  const float scale = 0.125f * 1.44269504088896340736f;
  v16h a0, a1;
  {
    const float* qr = qp + (size_t)(q0 + lr) * SROW + hi * 8;  // row M = lane%16
#pragma unroll
    for (int gq = 0; gq < 4; ++gq) {            // d groups 0,16,32,48
      float4 f0 = ((const float4*)(qr + gq * 16))[0];
      float4 f1 = ((const float4*)(qr + gq * 16))[1];
      v16h* dst = (gq < 2) ? &a0 : &a1;
      const int o = (gq & 1) * 8;
      (*dst)[o + 0] = (_Float16)(f0.x * scale);
      (*dst)[o + 1] = (_Float16)(f0.y * scale);
      (*dst)[o + 2] = (_Float16)(f0.z * scale);
      (*dst)[o + 3] = (_Float16)(f0.w * scale);
      (*dst)[o + 4] = (_Float16)(f1.x * scale);
      (*dst)[o + 5] = (_Float16)(f1.y * scale);
      (*dst)[o + 6] = (_Float16)(f1.z * scale);
      (*dst)[o + 7] = (_Float16)(f1.w * scale);
    }
  }

  // Constant all-ones B-fragment: P x Ones gives per-row sums replicated
  // across every output column -> row sums with zero cross-lane traffic.
  v16h ones;
#pragma unroll
  for (int j = 0; j < 16; ++j) ones[j] = (_Float16)1.0f;

  // online-softmax state (log2 domain; replicated across each 16-lane row)
  float m[8], l[8];
  v8f o0 = {}, o1 = {}, o2 = {}, o3 = {};
#pragma unroll
  for (int r = 0; r < 8; ++r) { m[r] = -INFINITY; l[r] = 0.0f; }

  int cc = 0;  // staged-chunk counter (double-buffer parity)
  for (int kb = 0; kb < NK; ++kb) {
    // union of the 16 layout rows in this group: uniform across the whole WG
    const int rowf = layout[(g * 16 + lr) * NK + kb];
    const bool any = (__ballot(rowf != 0) != 0ull);
    const int  own = __builtin_amdgcn_readfirstlane(layout[qb * NK + kb]);
    if (!any) continue;  // WG-uniform -> barrier-safe

    for (int c8 = 0; c8 < BLKK / 32; ++c8, ++cc) {
      const int kt  = kb * BLKK + c8 * 32;
      const int buf = cc & 1;

      // ---- cooperative staging: 512 lanes x 4 elements ----
      {
        const int srow = tid >> 4;           // key within chunk 0..31
        const int sc   = (tid & 15) << 2;    // d0 in {0,4,...,60}
        const float* kr = kp + (size_t)(kt + srow) * SROW + sc;
        float4 kf = *(const float4*)kr;
        union { _Float16 hh[4]; unsigned long long u; } pk;
        pk.hh[0] = (_Float16)kf.x; pk.hh[1] = (_Float16)kf.y;
        pk.hh[2] = (_Float16)kf.z; pk.hh[3] = (_Float16)kf.w;
        *(unsigned long long*)&klds[buf][srow * 64 + sc] = pk.u;   // ds_store_b64

        const float* vr = vp + (size_t)(kt + srow) * SROW + sc;
        float4 vf = *(const float4*)vr;
        vtld[buf][(sc + 0) * 32 + srow] = (_Float16)vf.x;          // transposed
        vtld[buf][(sc + 1) * 32 + srow] = (_Float16)vf.y;
        vtld[buf][(sc + 2) * 32 + srow] = (_Float16)vf.z;
        vtld[buf][(sc + 3) * 32 + srow] = (_Float16)vf.w;

        if (c8 + 1 < BLKK / 32) {            // prefetch next chunk
          __builtin_prefetch(kr + 32 * SROW, 0, 0);
          __builtin_prefetch(vr + 32 * SROW, 0, 0);
        }
      }
      __syncthreads();              // staging visible; also fences buffer reuse
      if (!own) continue;           // wave-uniform scalar skip, EXEC untouched

      // ---- S = Q * K^T for two 16-key tiles (B-frags: 2x ds_load_b128 each) ----
      v8f s[2];
#pragma unroll
      for (int t = 0; t < 2; ++t) {
        const _Float16* k0 = &klds[buf][(t * 16 + lr) * 64 + hi * 8];
        HV16 bk0, bk1;
        bk0.h[0] = *(const v8h*)(k0);       bk0.h[1] = *(const v8h*)(k0 + 16);
        bk1.h[0] = *(const v8h*)(k0 + 32);  bk1.h[1] = *(const v8h*)(k0 + 48);
        v8f acc = {};
        acc = __builtin_amdgcn_wmma_f32_16x16x32_f16(false, a0, false, bk0.v,
                                                     (short)0, acc, false, false);
        acc = __builtin_amdgcn_wmma_f32_16x16x32_f16(false, a1, false, bk1.v,
                                                     (short)0, acc, false, false);
        s[t] = acc;
      }

      // ---- online softmax over the 32 new keys (log2 domain, DPP max) ----
      float p0[8], p1[8];
#pragma unroll
      for (int r = 0; r < 8; ++r) {
        float cmax  = rowmax16(vmaxf(s[0][r], s[1][r]));
        float mnew  = vmaxf(m[r], cmax);
        float alpha = __builtin_amdgcn_exp2f(m[r] - mnew);
        p0[r] = __builtin_amdgcn_exp2f(s[0][r] - mnew);
        p1[r] = __builtin_amdgcn_exp2f(s[1][r] - mnew);
        m[r] = mnew;
        l[r] *= alpha;                 // row-sum contribution added below via WMMA
        o0[r] *= alpha; o1[r] *= alpha; o2[r] *= alpha; o3[r] *= alpha;
      }

      // ---- P (16x32) C-layout -> A-layout via per-wave LDS slice ----
      _Float16* pw = &plds[w][0];
#pragma unroll
      for (int r = 0; r < 8; ++r) {
        const int row = hi * 8 + r;
        pw[row * 32 +      lr] = (_Float16)p0[r];
        pw[row * 32 + 16 + lr] = (_Float16)p1[r];
      }
      HV16 pa;
      {
        const _Float16* pr = pw + lr * 32 + hi * 8;
        pa.h[0] = *(const v8h*)(pr);        // K = hi*8 + 0..7
        pa.h[1] = *(const v8h*)(pr + 16);   // K = 16 + hi*8 + 0..7
      }

      // ---- row sums: P x Ones -> every column holds rowsum(P) ----
      {
        v8f zsum = {};
        zsum = __builtin_amdgcn_wmma_f32_16x16x32_f16(false, pa.v, false, ones,
                                                      (short)0, zsum, false, false);
#pragma unroll
        for (int r = 0; r < 8; ++r) l[r] += zsum[r];
      }

      // ---- O += P * V over four 16-wide D tiles (V-frags: 2x ds_load_b128) ----
#pragma unroll
      for (int n = 0; n < 4; ++n) {
        const _Float16* vp0 = &vtld[buf][(n * 16 + lr) * 32 + hi * 8];
        HV16 bv;
        bv.h[0] = *(const v8h*)(vp0);
        bv.h[1] = *(const v8h*)(vp0 + 16);
        v8f* oacc = (n == 0) ? &o0 : (n == 1) ? &o1 : (n == 2) ? &o2 : &o3;
        *oacc = __builtin_amdgcn_wmma_f32_16x16x32_f16(false, pa.v, false, bv.v,
                                                       (short)0, *oacc, false, false);
      }
    }
  }

  // ---- epilogue: normalize, store fp32 ----
#pragma unroll
  for (int r = 0; r < 8; ++r) {
    const int row   = hi * 8 + r;
    const float inv = (l[r] > 0.0f) ? (1.0f / l[r]) : 0.0f;
    float* orow = out + ((size_t)b * S_ + q0 + row) * (H_ * D_) + h * D_;
    orow[ 0 + lr] = o0[r] * inv;
    orow[16 + lr] = o1[r] * inv;
    orow[32 + lr] = o2[r] * inv;
    orow[48 + lr] = o3[r] * inv;
  }
}

extern "C" void kernel_launch(void* const* d_in, const int* in_sizes, int n_in,
                              void* d_out, int out_size, void* d_ws, size_t ws_size,
                              hipStream_t stream) {
  const float* qkv    = (const float*)d_in[0];
  const int*   layout = (const int*)d_in[1];
  float*       out    = (float*)d_out;
  (void)in_sizes; (void)n_in; (void)out_size; (void)d_ws; (void)ws_size;

  dim3 grid(NQ / WAVES, H_, B_);   // (8, 16, 2) workgroups
  dim3 block(TPB);                 // 512 threads = 16 wave32
  fa_blocksparse_wmma_coop<<<grid, block, 0, stream>>>(qkv, layout, out);
}
